// MultiModalCrossAttentionMoE_5291399709241
// MI455X (gfx1250) — matrix operand order repaired and checked by reading, not compiled, not run on hardware
//
#include <hip/hip_runtime.h>
#include <hip/hip_bf16.h>
#include <math.h>

// ---------------------------------------------------------------------------
// MultiModal Cross-Attention MoE forward for gfx1250 (MI455X, wave32, WMMA)
// All heavy math runs through v_wmma_f32_16x16x32_bf16.
// ---------------------------------------------------------------------------

#define BDIM 4
#define TDIM 512
#define HDIM 768
#define EDIM 8
#define H4   3072
#define H2   1536
#define RROWS  2048   // B*T
#define R2ROWS 4096   // B*2T (fusion)

typedef __attribute__((ext_vector_type(16))) __bf16 v16bf;
typedef __attribute__((ext_vector_type(8)))  __bf16 v8bf;
typedef __attribute__((ext_vector_type(8)))  float  v8f;

// ---------------------------------------------------------------------------
// Generic tiled GEMM:  C = epilogue( A[M,K] @ Bw[K,N] + bias )
// flags bit0: exact GELU on (acc+bias); bit1: accumulate into C.
// residual  : added after gelu/scale (attention output residual)
// rowscale  : per-row multiplier with stride rsStride (MoE combine weights)
// Block tile 256x128, K-step 32, 8 waves (4x2), wave tile 64x64
// (4x4 WMMA frags -> 16 WMMA per 16 ds_load_b128 per K-step).
// M must be a multiple of 256 (all our Ms are 2048/4096); N/K arbitrary
// (K multiple of 32; N masked via block-uniform fast path).
// ---------------------------------------------------------------------------
__global__ __launch_bounds__(256) void gemm_bf16_kernel(
    const float* __restrict__ A, const float* __restrict__ Bw,
    const float* __restrict__ bias, const float* __restrict__ residual,
    const float* __restrict__ rowscale, int rsStride,
    float* __restrict__ C, int M, int N, int K, int flags)
{
    __shared__ __bf16 As[256 * 40];   // [row][k]  stride 40 (pad) -> 16B rows, conflict-free
    __shared__ __bf16 Bs[128 * 40];   // [col][k]  (B stored transposed)

    const int tid   = threadIdx.x;
    const int lane  = tid & 31;
    const int wave  = tid >> 5;
    const int wm    = wave & 3;        // 4 waves along M (64 rows each)
    const int wn    = wave >> 2;       // 2 waves along N (64 cols each)
    const int lrow  = lane & 15;
    const int khalf = (lane < 16) ? 0 : 8;
    const int m0    = blockIdx.y * 256;
    const int n0    = blockIdx.x * 128;
    const bool fullN = (n0 + 128 <= N);

    v8f acc[4][4];
    for (int m = 0; m < 4; ++m)
        for (int n = 0; n < 4; ++n)
            for (int j = 0; j < 8; ++j) acc[m][n][j] = 0.0f;

    // staging index precompute
    const int ar  = tid >> 3;          // 0..31  (A row within 32-row group)
    const int ac4 = (tid & 7) * 4;     // 0..28  (A k column, float4)
    const int bkk = tid >> 5;          // 0..7   (B k row within 8-row group)
    const int bn  = (tid & 31) * 4;    // 0..124 (B col, float4)

    for (int k0 = 0; k0 < K; k0 += 32) {
        // ---- stage A tile: 256 rows x 32 k. Issue all 8 loads, then cvt+store.
        {
            float4 f[8];
            for (int g = 0; g < 8; ++g)
                f[g] = *(const float4*)(A + (size_t)(m0 + g * 32 + ar) * K + k0 + ac4);
            for (int g = 0; g < 8; ++g) {
                __bf16* p = As + (g * 32 + ar) * 40 + ac4;
                p[0] = (__bf16)f[g].x; p[1] = (__bf16)f[g].y;
                p[2] = (__bf16)f[g].z; p[3] = (__bf16)f[g].w;
            }
        }
        // ---- stage B tile transposed: Bs[n][kk]. Block-uniform fast path.
        if (fullN) {
            float4 g[4];
            for (int k2 = 0; k2 < 4; ++k2)
                g[k2] = *(const float4*)(Bw + (size_t)(k0 + bkk + k2 * 8) * N + n0 + bn);
            for (int k2 = 0; k2 < 4; ++k2) {
                __bf16* q = Bs + bn * 40 + bkk + k2 * 8;
                q[0]   = (__bf16)g[k2].x;
                q[40]  = (__bf16)g[k2].y;
                q[80]  = (__bf16)g[k2].z;
                q[120] = (__bf16)g[k2].w;
            }
        } else {
            for (int k2 = 0; k2 < 4; ++k2) {
                const float* bp = Bw + (size_t)(k0 + bkk + k2 * 8) * N;
                const int gn = n0 + bn;
                float x0 = (gn     < N) ? bp[gn]     : 0.0f;
                float x1 = (gn + 1 < N) ? bp[gn + 1] : 0.0f;
                float x2 = (gn + 2 < N) ? bp[gn + 2] : 0.0f;
                float x3 = (gn + 3 < N) ? bp[gn + 3] : 0.0f;
                __bf16* q = Bs + bn * 40 + bkk + k2 * 8;
                q[0]   = (__bf16)x0;
                q[40]  = (__bf16)x1;
                q[80]  = (__bf16)x2;
                q[120] = (__bf16)x3;
            }
        }
        // ---- uniform-guarded prefetch of next A tile (no per-element branch)
        if (k0 + 32 < K) {
            for (int g = 0; g < 8; ++g)
                __builtin_prefetch(A + (size_t)(m0 + g * 32 + ar) * K + k0 + 32 + ac4, 0, 1);
        }
        __syncthreads();

        // ---- fragments ----
        // A frag (16x32 bf16): lanes 0-15 row M=lrow K{0..7,16..23}; lanes 16-31 K{8..15,24..31}
        v16bf af[4];
        for (int m = 0; m < 4; ++m) {
            const __bf16* p = As + (wm * 64 + m * 16 + lrow) * 40;
            const v8bf lo = *(const v8bf*)(p + khalf);
            const v8bf hi = *(const v8bf*)(p + 16 + khalf);
            for (int i = 0; i < 8; ++i) { af[m][i] = lo[i]; af[m][8 + i] = hi[i]; }
        }
        // B frag (32x16 bf16): lanes 0-15 col N=lrow K{0..7,16..23}; lanes 16-31 K{8..15,24..31}
        v16bf bf[4];
        for (int n = 0; n < 4; ++n) {
            const __bf16* p = Bs + (wn * 64 + n * 16 + lrow) * 40 + khalf;
            const v8bf lo = *(const v8bf*)(p);
            const v8bf hi = *(const v8bf*)(p + 16);
            for (int i = 0; i < 8; ++i) { bf[n][i] = lo[i]; bf[n][8 + i] = hi[i]; }
        }
        for (int m = 0; m < 4; ++m)
            for (int n = 0; n < 4; ++n)
                acc[m][n] = __builtin_amdgcn_wmma_f32_16x16x32_bf16(
                    false, af[m], false, bf[n], (short)0, acc[m][n], false, false);
        __syncthreads();
    }

    // ---- epilogue ----
    const bool do_gelu  = (flags & 1) != 0;
    const bool do_accum = (flags & 2) != 0;
    for (int m = 0; m < 4; ++m) {
        for (int n = 0; n < 4; ++n) {
            const int col = n0 + wn * 64 + n * 16 + lrow;
            if (col >= N) continue;
            const float bv = bias ? bias[col] : 0.0f;
            for (int j = 0; j < 8; ++j) {
                const int row = m0 + wm * 64 + m * 16 + j + ((lane < 16) ? 0 : 8);
                float v = acc[m][n][j] + bv;
                if (do_gelu) v = 0.5f * v * (1.0f + erff(v * 0.70710678118654752f));
                if (rowscale) v *= rowscale[(size_t)row * rsStride];
                const size_t idx = (size_t)row * N + col;
                if (residual) v += residual[idx];
                if (do_accum) C[idx] += v; else C[idx] = v;
            }
        }
    }
}

// ---------------------------------------------------------------------------
// Flash-style MHA core. Q,K,V,O are [Bn*Tq, HDIM] with head h at cols h*DH.
// Computes S^T = K @ Q^T per 16-key tile so softmax rows (queries) are
// per-lane; online softmax; O^T = V^T @ P^T. One wave per 16 queries,
// 8 waves (128 queries) per block; K/V tiles staged cooperatively.
// ---------------------------------------------------------------------------
template <int DH>
__global__ __launch_bounds__(256) void attn_kernel(
    const float* __restrict__ Q, const float* __restrict__ Kg,
    const float* __restrict__ Vg, float* __restrict__ Og,
    int nh, int Tq)
{
    constexpr int DH32 = DH / 32;
    constexpr int DH16 = DH / 16;
    constexpr int KSTR = DH + 8;       // bf16 elems; row stride stays 16B-aligned
    __shared__ __bf16 Ks[32 * KSTR];   // [key 0..31][d]
    __shared__ __bf16 Vs[DH * 40];     // [d][key]  (transposed, padded)

    const int qTiles = Tq / 128;
    const int qc = blockIdx.x % qTiles;
    const int h  = (blockIdx.x / qTiles) % nh;
    const int b  = blockIdx.x / (qTiles * nh);

    const int tid   = threadIdx.x;
    const int lane  = tid & 31;
    const int wave  = tid >> 5;
    const int lrow  = lane & 15;
    const int khalf = (lane < 16) ? 0 : 8;
    const size_t base = ((size_t)b * Tq) * HDIM + (size_t)h * DH;
    const int qb = qc * 128 + wave * 16;
    const float scale = rsqrtf((float)DH);

    // Preload Q^T B-fragments (loop invariant): lane<16 -> d = c*32+{0..7,16..23} of query lrow
    v16bf qf[DH32];
    {
        const float* qp = Q + base + (size_t)(qb + lrow) * HDIM + khalf;
        for (int c = 0; c < DH32; ++c)
            for (int i = 0; i < 16; i += 4) {
                const float4 f = *(const float4*)(qp + c * 32 + (i < 8 ? i : i + 8));
                qf[c][i]     = (__bf16)f.x; qf[c][i + 1] = (__bf16)f.y;
                qf[c][i + 2] = (__bf16)f.z; qf[c][i + 3] = (__bf16)f.w;
            }
    }

    float mrun = -3.0e38f, lrun = 0.0f;
    v8f oacc[DH16];
    for (int dt = 0; dt < DH16; ++dt)
        for (int j = 0; j < 8; ++j) oacc[dt][j] = 0.0f;

    for (int kt = 0; kt < Tq; kt += 32) {
        // stage K rows (32 keys x DH), f32 -> bf16; batch loads then stores
        {
            const int r  = tid >> 3;
            const int c4 = (tid & 7) * 4;
            float4 f[DH / 32];
            for (int d0 = 0; d0 < DH32; ++d0)
                f[d0] = *(const float4*)(Kg + base + (size_t)(kt + r) * HDIM + d0 * 32 + c4);
            for (int d0 = 0; d0 < DH32; ++d0) {
                __bf16* p = Ks + r * KSTR + d0 * 32 + c4;
                p[0] = (__bf16)f[d0].x; p[1] = (__bf16)f[d0].y;
                p[2] = (__bf16)f[d0].z; p[3] = (__bf16)f[d0].w;
            }
        }
        // stage V transposed: Vs[d][key]
        for (int i = tid; i < 32 * DH; i += 256) {
            const int key = i / DH, d = i % DH;
            Vs[d * 40 + key] = (__bf16)Vg[base + (size_t)(kt + key) * HDIM + d];
        }
        __syncthreads();

        // S^T tiles (keys 0-15 and 16-31) x (16 queries)
        v8f s0, s1;
        for (int j = 0; j < 8; ++j) { s0[j] = 0.0f; s1[j] = 0.0f; }
        for (int c = 0; c < DH32; ++c) {
            const __bf16* p0 = Ks + lrow * KSTR + c * 32 + khalf;
            const __bf16* p1 = Ks + (16 + lrow) * KSTR + c * 32 + khalf;
            v16bf a0, a1;
            {
                const v8bf l0 = *(const v8bf*)p0, h0 = *(const v8bf*)(p0 + 16);
                const v8bf l1 = *(const v8bf*)p1, h1 = *(const v8bf*)(p1 + 16);
                for (int i = 0; i < 8; ++i) {
                    a0[i] = l0[i]; a0[8 + i] = h0[i];
                    a1[i] = l1[i]; a1[8 + i] = h1[i];
                }
            }
            s0 = __builtin_amdgcn_wmma_f32_16x16x32_bf16(false, a0, false, qf[c], (short)0, s0, false, false);
            s1 = __builtin_amdgcn_wmma_f32_16x16x32_bf16(false, a1, false, qf[c], (short)0, s1, false, false);
        }

        // online softmax (query == lane%16; combine lane pair via xor-16)
        float lm = -3.0e38f;
        for (int j = 0; j < 8; ++j) {
            s0[j] *= scale; s1[j] *= scale;
            lm = fmaxf(lm, fmaxf(s0[j], s1[j]));
        }
        lm = fmaxf(lm, __shfl_xor(lm, 16));
        const float mnew = fmaxf(mrun, lm);
        const float fac  = __expf(mrun - mnew);
        float e0[8], e1[8], ls = 0.0f;
        for (int j = 0; j < 8; ++j) {
            e0[j] = __expf(s0[j] - mnew);
            e1[j] = __expf(s1[j] - mnew);
            ls += e0[j] + e1[j];
        }
        ls += __shfl_xor(ls, 16);
        lrun = lrun * fac + ls;
        mrun = mnew;
        for (int dt = 0; dt < DH16; ++dt)
            for (int j = 0; j < 8; ++j) oacc[dt][j] *= fac;

        // Build P^T B-fragment: lane<16 needs keys 0..15, lane>=16 keys 16..31
        v16bf pf;
        for (int j = 0; j < 8; ++j) {
            const float o0 = __shfl_xor(e0[j], 16);   // partner's keys 8..15 (tile0)
            const float o1 = __shfl_xor(e1[j], 16);   // partner's keys 16..23 (tile1)
            const float lo = (lane < 16) ? e0[j] : o1;
            const float hi = (lane < 16) ? o0    : e1[j];
            pf[j]     = (__bf16)lo;
            pf[8 + j] = (__bf16)hi;
        }

        // O^T += V^T @ P^T
        for (int dt = 0; dt < DH16; ++dt) {
            const __bf16* p = Vs + (dt * 16 + lrow) * 40 + ((lane < 16) ? 0 : 16);
            const v8bf lo = *(const v8bf*)p, hi = *(const v8bf*)(p + 8);
            v16bf a;
            for (int i = 0; i < 8; ++i) { a[i] = lo[i]; a[8 + i] = hi[i]; }
            oacc[dt] = __builtin_amdgcn_wmma_f32_16x16x32_bf16(false, a, false, pf, (short)0, oacc[dt], false, false);
        }
        __syncthreads();
    }

    const float inv = 1.0f / lrun;
    for (int dt = 0; dt < DH16; ++dt)
        for (int j = 0; j < 8; ++j) {
            const int d = dt * 16 + j + ((lane < 16) ? 0 : 8);
            Og[base + (size_t)(qb + lrow) * HDIM + d] = oacc[dt][j] * inv;
        }
}

// ---------------------------------------------------------------------------
// Small elementwise / routing / normalization kernels
// ---------------------------------------------------------------------------
__global__ void fill_zero_kernel(float* p, int n) {
    const int i = blockIdx.x * 256 + threadIdx.x;
    if (i < n) p[i] = 0.0f;
}

__global__ void concat2_kernel(const float* __restrict__ a, const float* __restrict__ b,
                               float* __restrict__ c, int R, int Hh) {
    const int i = blockIdx.x * 256 + threadIdx.x;
    if (i >= R * 2 * Hh) return;
    const int r = i / (2 * Hh), col = i % (2 * Hh);
    c[i] = (col < Hh) ? a[(size_t)r * Hh + col] : b[(size_t)r * Hh + (col - Hh)];
}

__global__ void interleave2_kernel(const float* __restrict__ a, const float* __restrict__ b,
                                   float* __restrict__ r, int Bn, int Tn, int Hh) {
    const int i = blockIdx.x * 256 + threadIdx.x;
    if (i >= Bn * Tn * 2 * Hh) return;
    const int col = i % Hh;
    const int row = i / Hh;
    const int m  = row & 1;
    const int t  = (row >> 1) % Tn;
    const int bb = row / (2 * Tn);
    const float* s = m ? b : a;
    r[i] = s[((size_t)bb * Tn + t) * Hh + col];
}

__global__ void router_kernel(const float* __restrict__ logits,
                              float* __restrict__ cwv, float* __restrict__ cwt, int R) {
    const int t = blockIdx.x * 256 + threadIdx.x;
    if (t >= R) return;
    float p[16];
    float mx = -3.0e38f;
    for (int i = 0; i < 16; ++i) { p[i] = logits[(size_t)t * 16 + i]; mx = fmaxf(mx, p[i]); }
    float s = 0.0f;
    for (int i = 0; i < 16; ++i) { p[i] = __expf(p[i] - mx); s += p[i]; }
    const float invs = 1.0f / s;
    for (int i = 0; i < 16; ++i) p[i] *= invs;
    for (int sl = 0; sl < 2; ++sl) {
        const float* q = p + sl * 8;
        int i1 = 0, i2 = 0;
        float v1 = -1.0f, v2 = -1.0f;
        for (int i = 0; i < 8; ++i) {
            const float x = q[i];
            if (x > v1)      { v2 = v1; i2 = i1; v1 = x; i1 = i; }
            else if (x > v2) { v2 = x;  i2 = i; }
        }
        float den = v1 + v2;
        den = (den < 1e-12f) ? 1e-12f : den;
        float* cw = sl ? (cwt + (size_t)t * 8) : (cwv + (size_t)t * 8);
        for (int i = 0; i < 8; ++i) cw[i] = 0.0f;
        cw[i1] = v1 / den;
        cw[i2] += v2 / den;
    }
}

// mean over the two modality rows + LayerNorm -> output [B*T, H]
__global__ __launch_bounds__(256) void mean_ln_kernel(
    const float* __restrict__ fp, const float* __restrict__ g,
    const float* __restrict__ bta, float* __restrict__ out, int Tn, int Hh) {
    const int token = blockIdx.x;
    const int bb = token / Tn, t = token % Tn;
    const float* r0 = fp + ((size_t)bb * 2 * Tn + 2 * t) * Hh;
    const float* r1 = r0 + Hh;
    float x[3], s = 0.0f, ss = 0.0f;
    for (int e = 0; e < 3; ++e) {
        const int c = threadIdx.x + e * 256;
        const float v = 0.5f * (r0[c] + r1[c]);
        x[e] = v; s += v; ss += v * v;
    }
    __shared__ float rs[256], rq[256];
    rs[threadIdx.x] = s; rq[threadIdx.x] = ss;
    __syncthreads();
    for (int st = 128; st > 0; st >>= 1) {
        if (threadIdx.x < st) {
            rs[threadIdx.x] += rs[threadIdx.x + st];
            rq[threadIdx.x] += rq[threadIdx.x + st];
        }
        __syncthreads();
    }
    const float mean = rs[0] / (float)Hh;
    const float var  = rq[0] / (float)Hh - mean * mean;
    const float inv  = rsqrtf(var + 1e-5f);
    float* o = out + (size_t)token * Hh;
    for (int e = 0; e < 3; ++e) {
        const int c = threadIdx.x + e * 256;
        o[c] = (x[e] - mean) * inv * g[c] + bta[c];
    }
}

// ---------------------------------------------------------------------------
// Host orchestration
// ---------------------------------------------------------------------------
static void gemm(hipStream_t st, const float* A, const float* Bw, const float* bias,
                 const float* residual, const float* rowscale, int rsStride,
                 float* C, int M, int N, int K, int flags) {
    dim3 grid((N + 127) / 128, M / 256);
    gemm_bf16_kernel<<<grid, 256, 0, st>>>(A, Bw, bias, residual, rowscale, rsStride,
                                           C, M, N, K, flags);
}

extern "C" void kernel_launch(void* const* d_in, const int* in_sizes, int n_in,
                              void* d_out, int out_size, void* d_ws, size_t ws_size,
                              hipStream_t stream) {
    (void)in_sizes; (void)n_in; (void)out_size; (void)ws_size;
    const float** in = (const float**)d_in;
    const float *xv = in[0], *xt = in[1];
    const float *Wp_v = in[2], *bp_v = in[3], *Wp_t = in[4], *bp_t = in[5];
    const float *vtWq = in[6],  *vtWk = in[7],  *vtWv = in[8],  *vtWo = in[9];
    const float *vtbq = in[10], *vtbk = in[11], *vtbv = in[12], *vtbo = in[13];
    const float *tvWq = in[14], *tvWk = in[15], *tvWv = in[16], *tvWo = in[17];
    const float *tvbq = in[18], *tvbk = in[19], *tvbv = in[20], *tvbo = in[21];
    const float *fuWq = in[22], *fuWk = in[23], *fuWv = in[24], *fuWo = in[25];
    const float *fubq = in[26], *fubk = in[27], *fubv = in[28], *fubo = in[29];
    const float *Wr1 = in[30], *br1 = in[31], *Wr2 = in[32], *br2 = in[33];
    const float *exvW1 = in[34], *exvb1 = in[35], *exvW2 = in[36], *exvb2 = in[37];
    const float *extW1 = in[38], *extb1 = in[39], *extW2 = in[40], *extb2 = in[41];
    const float *fug = in[42], *fub = in[43];
    float* out = (float*)d_out;

    // workspace layout (floats)
    float* ws = (float*)d_ws;
    size_t off = 0;
    const size_t RH  = (size_t)RROWS * HDIM;
    const size_t R2H = (size_t)R2ROWS * HDIM;
    float* pv    = ws + off; off += RH;
    float* pt    = ws + off; off += RH;
    float* av    = ws + off; off += RH;
    float* at    = ws + off; off += RH;
    float* ov    = ws + off; off += RH;
    float* ot    = ws + off; off += RH;
    float* qb    = ws + off; off += R2H;
    float* kb    = ws + off; off += R2H;
    float* vb    = ws + off; off += R2H;
    float* ab    = ws + off; off += R2H;   // attention output (pre-proj)
    float* rb    = ws + off; off += R2H;   // stacked fusion input
    float* fpb   = ws + off; off += R2H;   // fusion projected output
    float* cbuf  = ws + off; off += (size_t)RROWS * H2;
    float* g1    = ws + off; off += (size_t)RROWS * H2;
    float* hexp  = ws + off; off += (size_t)RROWS * H4;
    float* lgt   = ws + off; off += (size_t)RROWS * 16;
    float* cwv   = ws + off; off += (size_t)RROWS * 8;
    float* cwt   = ws + off; off += (size_t)RROWS * 8;

    const int GELU = 1, ACCUM = 2;

    // 1) modality projections
    gemm(stream, xv, Wp_v, bp_v, nullptr, nullptr, 0, pv, RROWS, HDIM, HDIM, 0);
    gemm(stream, xt, Wp_t, bp_t, nullptr, nullptr, 0, pt, RROWS, HDIM, HDIM, 0);

    // 2) cross-attention vision<-text (q from pv, k/v from pt), residual pv
    gemm(stream, pv, vtWq, vtbq, nullptr, nullptr, 0, qb, RROWS, HDIM, HDIM, 0);
    gemm(stream, pt, vtWk, vtbk, nullptr, nullptr, 0, kb, RROWS, HDIM, HDIM, 0);
    gemm(stream, pt, vtWv, vtbv, nullptr, nullptr, 0, vb, RROWS, HDIM, HDIM, 0);
    attn_kernel<64><<<BDIM * 12 * (TDIM / 128), 256, 0, stream>>>(qb, kb, vb, ab, 12, TDIM);
    gemm(stream, ab, vtWo, vtbo, pv, nullptr, 0, av, RROWS, HDIM, HDIM, 0);

    // 3) cross-attention text<-vision (q from pt, k/v from pv), residual pt
    gemm(stream, pt, tvWq, tvbq, nullptr, nullptr, 0, qb, RROWS, HDIM, HDIM, 0);
    gemm(stream, pv, tvWk, tvbk, nullptr, nullptr, 0, kb, RROWS, HDIM, HDIM, 0);
    gemm(stream, pv, tvWv, tvbv, nullptr, nullptr, 0, vb, RROWS, HDIM, HDIM, 0);
    attn_kernel<64><<<BDIM * 12 * (TDIM / 128), 256, 0, stream>>>(qb, kb, vb, ab, 12, TDIM);
    gemm(stream, ab, tvWo, tvbo, pt, nullptr, 0, at, RROWS, HDIM, HDIM, 0);

    // 4) router: c = [av, at]; logits = gelu(c@Wr1+br1)@Wr2+br2; softmax+top2
    {
        const int n = RROWS * 2 * HDIM;
        concat2_kernel<<<(n + 255) / 256, 256, 0, stream>>>(av, at, cbuf, RROWS, HDIM);
    }
    gemm(stream, cbuf, Wr1, br1, nullptr, nullptr, 0, g1, RROWS, H2, H2, GELU);
    gemm(stream, g1, Wr2, br2, nullptr, nullptr, 0, lgt, RROWS, 16, H2, 0);
    router_kernel<<<(RROWS + 255) / 256, 256, 0, stream>>>(lgt, cwv, cwt, RROWS);

    // 5) dense top-2 MoE per modality (combine weights are 0 for non-selected)
    {
        const int n = (int)RH;
        fill_zero_kernel<<<(n + 255) / 256, 256, 0, stream>>>(ov, n);
        fill_zero_kernel<<<(n + 255) / 256, 256, 0, stream>>>(ot, n);
    }
    for (int e = 0; e < EDIM; ++e) {
        gemm(stream, av, exvW1 + (size_t)e * HDIM * H4, exvb1 + (size_t)e * H4,
             nullptr, nullptr, 0, hexp, RROWS, H4, HDIM, GELU);
        gemm(stream, hexp, exvW2 + (size_t)e * H4 * HDIM, exvb2 + (size_t)e * HDIM,
             nullptr, cwv + e, 8, ov, RROWS, HDIM, H4, ACCUM);
    }
    for (int e = 0; e < EDIM; ++e) {
        gemm(stream, at, extW1 + (size_t)e * HDIM * H4, extb1 + (size_t)e * H4,
             nullptr, nullptr, 0, hexp, RROWS, H4, HDIM, GELU);
        gemm(stream, hexp, extW2 + (size_t)e * H4 * HDIM, extb2 + (size_t)e * HDIM,
             nullptr, cwt + e, 8, ot, RROWS, HDIM, H4, ACCUM);
    }

    // 6) fusion self-attention over stacked modalities [B, 2T, H]
    {
        const int n = (int)R2H;
        interleave2_kernel<<<(n + 255) / 256, 256, 0, stream>>>(ov, ot, rb, BDIM, TDIM, HDIM);
    }
    gemm(stream, rb, fuWq, fubq, nullptr, nullptr, 0, qb, R2ROWS, HDIM, HDIM, 0);
    gemm(stream, rb, fuWk, fubk, nullptr, nullptr, 0, kb, R2ROWS, HDIM, HDIM, 0);
    gemm(stream, rb, fuWv, fubv, nullptr, nullptr, 0, vb, R2ROWS, HDIM, HDIM, 0);
    attn_kernel<96><<<BDIM * 8 * (2 * TDIM / 128), 256, 0, stream>>>(qb, kb, vb, ab, 8, 2 * TDIM);
    gemm(stream, ab, fuWo, fubo, nullptr, nullptr, 0, fpb, R2ROWS, HDIM, HDIM, 0);

    // 7) mean over modalities + LayerNorm -> d_out [B*T, H]
    mean_ln_kernel<<<RROWS, 256, 0, stream>>>(fpb, fug, fub, out, TDIM, HDIM);
}
